// GFMEncoder_18287970747040
// MI455X (gfx1250) — matrix operand order repaired and checked by reading, not compiled, not run on hardware
//
#include <hip/hip_runtime.h>

typedef __attribute__((ext_vector_type(16))) __bf16 bf16x16;
typedef __attribute__((ext_vector_type(8)))  __bf16 bf16x8;
typedef __attribute__((ext_vector_type(8)))  float  f32x8;

#define D_MODEL   128
#define NHEAD     8
#define D_HEAD    16
#define NUM_KEYS  8
#define NUM_ETYPES 4

// ---------------- elementwise helpers ----------------

__global__ void k_convert_bf16(const float* __restrict__ in, __bf16* __restrict__ out, int n) {
  int i = blockIdx.x * blockDim.x + threadIdx.x;
  if (i < n) out[i] = (__bf16)in[i];
}

__global__ void k_gather_embed(const float* __restrict__ emb, const int* __restrict__ vids,
                               float* __restrict__ feat, __bf16* __restrict__ featb, int total) {
  int i = blockIdx.x * blockDim.x + threadIdx.x;
  if (i >= total) return;
  int n = i >> 7, c = i & 127;
  float v = emb[(size_t)vids[n] * D_MODEL + c];
  feat[i] = v;
  featb[i] = (__bf16)v;
}

__global__ void k_init_accum(float* __restrict__ m, float* __restrict__ z,
                             float* __restrict__ agg, int nN) {
  int i = blockIdx.x * blockDim.x + threadIdx.x;
  if (i < nN * D_MODEL) agg[i] = 0.f;
  if (i < nN * NHEAD) { m[i] = -3.402823466e38f; z[i] = 0.f; }
}

// ---------------- WMMA GEMM: C[row, col] = sum_d A[row,d] * W[col,d] ----------------
// A: [nrows,128] bf16 row-major.  W: [ncols,128] bf16 row-major (weight rows = output cols).
// SELECT=false: store bf16 C into outB [nrows, ncols]   (k/v projections)
// SELECT=true : ncols = NUM_KEYS*128; store f32 only where key_ids[row]==col>>7 into outF [nrows,128]

__device__ inline bf16x16 load_a16(const __bf16* p) {
  bf16x8 lo = *(const bf16x8*)p;
  bf16x8 hi = *(const bf16x8*)(p + 8);
  bf16x16 r;
#pragma unroll
  for (int i = 0; i < 8; ++i) { r[i] = lo[i]; r[i + 8] = hi[i]; }
  return r;
}

template <bool SELECT>
__global__ void k_gemm_wmma(const __bf16* __restrict__ A,
                            const __bf16* __restrict__ W,
                            __bf16* __restrict__ outB,
                            float* __restrict__ outF,
                            const int* __restrict__ key_ids,
                            int nrows, int ncols) {
  const int tm    = blockIdx.x * 16;   // row tile
  const int tn    = blockIdx.y * 64;   // col tile (4 x 16)
  const int lane  = threadIdx.x & 31;
  const int r     = lane & 15;
  const int khalf = lane >> 4;

  int arow = tm + r;
  if (arow >= nrows) arow = nrows - 1;
  const __bf16* ap = A + (size_t)arow * D_MODEL + khalf * 16;

  f32x8 acc0 = {}, acc1 = {}, acc2 = {}, acc3 = {};
#pragma unroll
  for (int kb = 0; kb < D_MODEL; kb += 32) {
    bf16x16 a = load_a16(ap + kb);
    const __bf16* wp = W + (size_t)(tn + r) * D_MODEL + khalf * 16 + kb;
    bf16x16 b0 = *(const bf16x16*)(wp);
    bf16x16 b1 = *(const bf16x16*)(wp + 16 * D_MODEL);
    bf16x16 b2 = *(const bf16x16*)(wp + 32 * D_MODEL);
    bf16x16 b3 = *(const bf16x16*)(wp + 48 * D_MODEL);
    acc0 = __builtin_amdgcn_wmma_f32_16x16x32_bf16(false, a, false, b0, (short)0, acc0, false, false);
    acc1 = __builtin_amdgcn_wmma_f32_16x16x32_bf16(false, a, false, b1, (short)0, acc1, false, false);
    acc2 = __builtin_amdgcn_wmma_f32_16x16x32_bf16(false, a, false, b2, (short)0, acc2, false, false);
    acc3 = __builtin_amdgcn_wmma_f32_16x16x32_bf16(false, a, false, b3, (short)0, acc3, false, false);
  }
  f32x8 accs[4] = {acc0, acc1, acc2, acc3};
#pragma unroll
  for (int j = 0; j < 4; ++j) {
    int col = tn + j * 16 + r;
#pragma unroll
    for (int v = 0; v < 8; ++v) {
      int row = tm + khalf * 8 + v;
      if (row >= nrows) continue;
      float val = accs[j][v];
      if constexpr (SELECT) {
        if (key_ids[row] == (col >> 7))
          outF[(size_t)row * D_MODEL + (col & 127)] = val;
      } else {
        outB[(size_t)row * ncols + col] = (__bf16)val;
      }
    }
  }
}

// ---------------- edge softmax passes ----------------

__device__ inline void atomicMaxF(float* a, float v) {
  if (v >= 0.f) atomicMax((int*)a, __float_as_int(v));
  else          atomicMin((unsigned int*)a, __float_as_uint(v));
}

__global__ void k_edge_scores(const float* __restrict__ qsel, const __bf16* __restrict__ kall,
                              const int* __restrict__ src, const int* __restrict__ dst,
                              const int* __restrict__ et, float* __restrict__ scores,
                              float* __restrict__ mmax, int nE) {
  int i = blockIdx.x * blockDim.x + threadIdx.x;
  if (i >= nE * NHEAD) return;
  int e = i >> 3, h = i & 7;
  int s = src[e], d = dst[e], t = et[e];
  const float*  qp = qsel + (size_t)d * D_MODEL + h * D_HEAD;
  const __bf16* kp = kall + (size_t)s * (NUM_ETYPES * D_MODEL) + t * D_MODEL + h * D_HEAD;
  float acc = 0.f;
#pragma unroll
  for (int f = 0; f < D_HEAD; ++f) acc += qp[f] * (float)kp[f];
  acc *= 0.25f;  // 1/sqrt(D_HEAD)
  scores[i] = acc;
  atomicMaxF(&mmax[d * NHEAD + h], acc);
}

__global__ void k_edge_accum(const float* __restrict__ scores, const float* __restrict__ mmax,
                             float* __restrict__ z, const __bf16* __restrict__ vall,
                             const int* __restrict__ src, const int* __restrict__ dst,
                             const int* __restrict__ et, float* __restrict__ agg, int nE) {
  int i = blockIdx.x * blockDim.x + threadIdx.x;
  if (i >= nE * NHEAD) return;
  int e = i >> 3, h = i & 7;
  int s = src[e], d = dst[e], t = et[e];
  float w = __expf(scores[i] - mmax[d * NHEAD + h]);
  atomicAdd(&z[d * NHEAD + h], w);
  const __bf16* vp = vall + (size_t)s * (NUM_ETYPES * D_MODEL) + t * D_MODEL + h * D_HEAD;
  float* ap = agg + (size_t)d * D_MODEL + h * D_HEAD;
#pragma unroll
  for (int f = 0; f < D_HEAD; ++f) atomicAdd(&ap[f], w * (float)vp[f]);
}

__global__ void k_norm_div(const float* __restrict__ agg, const float* __restrict__ z,
                           __bf16* __restrict__ aggb, int total) {
  int i = blockIdx.x * blockDim.x + threadIdx.x;
  if (i >= total) return;
  int n = i >> 7, h = (i >> 4) & 7;
  aggb[i] = (__bf16)(agg[i] / (z[n * NHEAD + h] + 1e-9f));
}

// ---------------- relu -> layernorm -> residual (one wave32 per node) ----------------

__global__ void k_ln_residual(const float* __restrict__ hsel, const float* __restrict__ g,
                              const float* __restrict__ b, float* __restrict__ feat,
                              __bf16* __restrict__ featb, int nN) {
  int tid  = blockIdx.x * blockDim.x + threadIdx.x;
  int node = tid >> 5;
  int lane = tid & 31;
  if (node >= nN) return;  // whole wave exits together (node uniform per wave)
  float x[4];
  float sum = 0.f;
#pragma unroll
  for (int i = 0; i < 4; ++i) {
    float v = hsel[(size_t)node * D_MODEL + lane + i * 32];
    v = v > 0.f ? v : 0.f;  // relu before layernorm
    x[i] = v; sum += v;
  }
#pragma unroll
  for (int off = 16; off > 0; off >>= 1) sum += __shfl_xor(sum, off, 32);
  float mu = sum * (1.f / 128.f);
  float sq = 0.f;
#pragma unroll
  for (int i = 0; i < 4; ++i) { float dv = x[i] - mu; sq += dv * dv; }
#pragma unroll
  for (int off = 16; off > 0; off >>= 1) sq += __shfl_xor(sq, off, 32);
  float rs = rsqrtf(sq * (1.f / 128.f) + 1e-5f);
#pragma unroll
  for (int i = 0; i < 4; ++i) {
    int c = lane + i * 32;
    float y  = (x[i] - mu) * rs * g[c] + b[c];
    float nf = y + feat[(size_t)node * D_MODEL + c];  // block residual
    feat[(size_t)node * D_MODEL + c]  = nf;
    featb[(size_t)node * D_MODEL + c] = (__bf16)nf;
  }
}

// ---------------- target readout (mean over blocks) ----------------

__global__ void k_target_readout(const float* __restrict__ feat, const float* __restrict__ tw,
                                 const int* __restrict__ tgt, const int* __restrict__ key_ids,
                                 float* __restrict__ out, int nT, int first) {
  int bi = blockIdx.x * blockDim.x + threadIdx.x;
  if (bi >= nT) return;
  int n = tgt[bi];
  int k = key_ids[n];
  const float* w = tw + (size_t)k * D_MODEL;
  const float* f = feat + (size_t)n * D_MODEL;
  float s = 0.f;
  for (int c = 0; c < D_MODEL; ++c) s += f[c] * w[c];
  s *= 0.5f;  // mean over NUM_BLOCKS=2
  if (first) out[bi] = s; else out[bi] += s;
}

// ---------------- host orchestration ----------------

extern "C" void kernel_launch(void* const* d_in, const int* in_sizes, int n_in,
                              void* d_out, int out_size, void* d_ws, size_t ws_size,
                              hipStream_t stream) {
  const float* emb  = (const float*)d_in[0];
  const float* qw   = (const float*)d_in[1];   // [2,8,8,16,128]
  const float* kw   = (const float*)d_in[2];   // [2,4,8,16,128]
  const float* vw   = (const float*)d_in[3];   // [2,4,8,16,128]
  const float* nw   = (const float*)d_in[4];   // [2,8,8,16,128]
  const float* tw   = (const float*)d_in[5];   // [2,8,8,16]
  const float* lng  = (const float*)d_in[6];   // [2,128]
  const float* lnb  = (const float*)d_in[7];   // [2,128]
  const int* vids   = (const int*)d_in[8];
  const int* kids   = (const int*)d_in[9];
  const int* src    = (const int*)d_in[10];
  const int* dst    = (const int*)d_in[11];
  const int* et     = (const int*)d_in[12];
  const int* tgt    = (const int*)d_in[13];
  float* out = (float*)d_out;

  const int N  = in_sizes[8];
  const int E  = in_sizes[10];
  const int nT = in_sizes[13];
  (void)n_in; (void)out_size; (void)ws_size;

  char* p = (char*)d_ws;
  auto carve = [&](size_t bytes) -> char* {
    char* r = p;
    p += (bytes + 255) & ~(size_t)255;
    return r;
  };
  float*  feat   = (float*)carve((size_t)N * 128 * 4);
  __bf16* featb  = (__bf16*)carve((size_t)N * 128 * 2);
  float*  qsel   = (float*)carve((size_t)N * 128 * 4);
  __bf16* kall   = (__bf16*)carve((size_t)N * 512 * 2);
  __bf16* vall   = (__bf16*)carve((size_t)N * 512 * 2);
  float*  scores = (float*)carve((size_t)E * 8 * 4);
  float*  mmax   = (float*)carve((size_t)N * 8 * 4);
  float*  zsum   = (float*)carve((size_t)N * 8 * 4);
  float*  agg    = (float*)carve((size_t)N * 128 * 4);
  __bf16* aggb   = (__bf16*)carve((size_t)N * 128 * 2);
  float*  hsel   = (float*)carve((size_t)N * 128 * 4);
  __bf16* qwb    = (__bf16*)carve((size_t)2 * 1024 * 128 * 2);
  __bf16* kwb    = (__bf16*)carve((size_t)2 * 512 * 128 * 2);
  __bf16* vwb    = (__bf16*)carve((size_t)2 * 512 * 128 * 2);
  __bf16* nwb    = (__bf16*)carve((size_t)2 * 1024 * 128 * 2);

  const int TB = 256;
  k_convert_bf16<<<(2 * 1024 * 128 + TB - 1) / TB, TB, 0, stream>>>(qw, qwb, 2 * 1024 * 128);
  k_convert_bf16<<<(2 * 512 * 128 + TB - 1) / TB, TB, 0, stream>>>(kw, kwb, 2 * 512 * 128);
  k_convert_bf16<<<(2 * 512 * 128 + TB - 1) / TB, TB, 0, stream>>>(vw, vwb, 2 * 512 * 128);
  k_convert_bf16<<<(2 * 1024 * 128 + TB - 1) / TB, TB, 0, stream>>>(nw, nwb, 2 * 1024 * 128);
  k_gather_embed<<<(N * 128 + TB - 1) / TB, TB, 0, stream>>>(emb, vids, feat, featb, N * 128);

  const int mtiles = (N + 15) / 16;
  for (int blk = 0; blk < 2; ++blk) {
    // k/v projections for all etypes: [N, 4*128] bf16
    k_gemm_wmma<false><<<dim3(mtiles, 512 / 64), 32, 0, stream>>>(
        featb, kwb + (size_t)blk * 512 * 128, kall, nullptr, nullptr, N, 512);
    k_gemm_wmma<false><<<dim3(mtiles, 512 / 64), 32, 0, stream>>>(
        featb, vwb + (size_t)blk * 512 * 128, vall, nullptr, nullptr, N, 512);
    // q for all 8 key types, keep only the node's own key
    k_gemm_wmma<true><<<dim3(mtiles, 1024 / 64), 32, 0, stream>>>(
        featb, qwb + (size_t)blk * 1024 * 128, nullptr, qsel, kids, N, 1024);

    k_init_accum<<<(N * 128 + TB - 1) / TB, TB, 0, stream>>>(mmax, zsum, agg, N);
    k_edge_scores<<<(E * 8 + TB - 1) / TB, TB, 0, stream>>>(qsel, kall, src, dst, et, scores, mmax, E);
    k_edge_accum<<<(E * 8 + TB - 1) / TB, TB, 0, stream>>>(scores, mmax, zsum, vall, src, dst, et, agg, E);
    k_norm_div<<<(N * 128 + TB - 1) / TB, TB, 0, stream>>>(agg, zsum, aggb, N * 128);

    // node projection (all 8 keys, select own key)
    k_gemm_wmma<true><<<dim3(mtiles, 1024 / 64), 32, 0, stream>>>(
        aggb, nwb + (size_t)blk * 1024 * 128, nullptr, hsel, kids, N, 1024);

    k_ln_residual<<<(N * 32 + TB - 1) / TB, TB, 0, stream>>>(
        hsel, lng + blk * 128, lnb + blk * 128, feat, featb, N);

    k_target_readout<<<(nT + 127) / 128, 128, 0, stream>>>(
        feat, tw + (size_t)blk * NUM_KEYS * 128, tgt, kids, out, nT, blk == 0);
  }
}
